// MultiHeadAttention_27702539059554
// MI455X (gfx1250) — matrix operand order repaired and checked by reading, not compiled, not run on hardware
//
#include <hip/hip_runtime.h>
#include <hip/hip_bf16.h>

// ---------------------------------------------------------------------------
// MI455X (gfx1250) multi-head attention, bf16 WMMA pipeline.
//  - all matmuls on v_wmma_f32_16x16x32_bf16 (f32 accumulate)
//  - flash-attention inner loop (scores never reach HBM)
//  - V projection written pre-transposed [B,H,HEAD_DIM,S] so attention needs
//    no LDS transpose
//  - tile staging via global_load_async_to_lds_b128 (ASYNCcnt) when the
//    toolchain exposes it, vector-copy fallback otherwise
// ---------------------------------------------------------------------------

typedef __attribute__((ext_vector_type(16))) __bf16 bf16x16;
typedef __attribute__((ext_vector_type(8)))  float  floatx8;
typedef __attribute__((ext_vector_type(4)))  int    intx4;
typedef __attribute__((ext_vector_type(4)))  float  floatx4;

#define D_MODEL   1024
#define NUM_HEADS 16
#define HEAD_DIM  64
#define SEQ       2048
#define BATCH     2

#define ASG __attribute__((address_space(1)))
#define ASL __attribute__((address_space(3)))

#if __has_builtin(__builtin_amdgcn_global_load_async_to_lds_b128)
#define HAS_ASYNC_LDS 1
#else
#define HAS_ASYNC_LDS 0
#endif

// 16-byte global -> LDS tile copy. Async (ASYNCcnt) when available.
// Builtin signature (from clang diagnostic): param 0 is a pointer to
// int __vector(4) in the global AS; param 1 mirrors it in LDS AS.
__device__ __forceinline__ void stage16(const void* gsrc, void* ldst) {
#if HAS_ASYNC_LDS
    __builtin_amdgcn_global_load_async_to_lds_b128(
        (ASG intx4*)(unsigned long long)(size_t)gsrc,
        (ASL intx4*)(unsigned int)(size_t)ldst, 0, 0);
#else
    *(intx4*)ldst = *(const intx4*)gsrc;
#endif
}

__device__ __forceinline__ void wait_stage() {
#if HAS_ASYNC_LDS
    asm volatile("s_wait_asynccnt 0x0" ::: "memory");
#endif
}

__device__ __forceinline__ unsigned short f2bf(float f) {
    unsigned int u = __builtin_bit_cast(unsigned int, f);
    u += 0x7FFFu + ((u >> 16) & 1u);          // round-to-nearest-even
    return (unsigned short)(u >> 16);
}

union FragA { intx4 q[2]; bf16x16 v; };       // 32B: 16 bf16 operand fragment

// ---------------------------------------------------------------------------
// f32 -> bf16 conversion (vectorized 4-wide)
// ---------------------------------------------------------------------------
__global__ void cvt_f32_bf16(const float* __restrict__ src,
                             unsigned short* __restrict__ dst, int n4) {
    int i = blockIdx.x * blockDim.x + threadIdx.x;
    if (i >= n4) return;
    floatx4 f = ((const floatx4*)src)[i];
    unsigned int lo = (unsigned int)f2bf(f.x) | ((unsigned int)f2bf(f.y) << 16);
    unsigned int hi = (unsigned int)f2bf(f.z) | ((unsigned int)f2bf(f.w) << 16);
    ((unsigned int*)dst)[2*i]   = lo;
    ((unsigned int*)dst)[2*i+1] = hi;
}

// ---------------------------------------------------------------------------
// GEMM: Y[M,N] = A[M,K] (bf16 row-major) @ W[N,K]^T (bf16, row = out feature)
// Block tile 128x64, 4 waves, wave tile 32x64, BK=32.
// OUT_MODE 1: bf16 scattered to [B,H,S,64]       (Q, K)
// OUT_MODE 3: bf16 scattered to [B,H,64,S]       (V, pre-transposed)
// OUT_MODE 2: f32 + bias row-major               (final projection)
// ---------------------------------------------------------------------------
template<int OUT_MODE>
__global__ void __launch_bounds__(128)
gemm_wmma_bf16(const unsigned short* __restrict__ A,
               const unsigned short* __restrict__ W,
               unsigned short* __restrict__ Ybf,
               float* __restrict__ Yf,
               const float* __restrict__ bias,
               int M, int N, int K)
{
    constexpr int BM = 128, BN = 64, BK = 32, PAD = 8;
    __shared__ unsigned short As[BM][BK + PAD];   // [m][k]
    __shared__ unsigned short Bs[BN][BK + PAD];   // [n][k]

    const int t    = threadIdx.x;
    const int lane = t & 31;
    const int wave = t >> 5;
    const int m0   = blockIdx.x * BM;
    const int n0   = blockIdx.y * BN;
    const int wrow = wave * 32;

    floatx8 acc[2][4];
    #pragma unroll
    for (int i = 0; i < 2; ++i)
        #pragma unroll
        for (int j = 0; j < 4; ++j) acc[i][j] = 0.0f;

    for (int k0 = 0; k0 < K; k0 += BK) {
        // Stage A tile: 128 rows x 32 bf16 = 512 x 16B chunks, 4 per thread
        #pragma unroll
        for (int i = 0; i < 4; ++i) {
            int idx = t + 128 * i;
            int r = idx >> 2, c = idx & 3;
            stage16(A + (size_t)(m0 + r) * K + k0 + c * 8, &As[r][c * 8]);
        }
        // Stage B tile: 64 rows x 32 bf16 = 256 x 16B chunks, 2 per thread
        #pragma unroll
        for (int i = 0; i < 2; ++i) {
            int idx = t + 128 * i;
            int r = idx >> 2, c = idx & 3;
            stage16(W + (size_t)(n0 + r) * K + k0 + c * 8, &Bs[r][c * 8]);
        }
        if (k0 + BK < K)   // pull next A tile toward GL2 (global_prefetch_b8)
            __builtin_prefetch(A + (size_t)(m0 + (t >> 2)) * K + k0 + BK, 0, 0);
        wait_stage();
        __syncthreads();

        // A fragment: lane m = l&15, two 8-elem K runs selected by l>>4
        FragA a[2], b[4];
        const int kb = (lane >> 4) * 8;
        #pragma unroll
        for (int mt = 0; mt < 2; ++mt) {
            int m = wrow + 16 * mt + (lane & 15);
            a[mt].q[0] = *(const intx4*)&As[m][kb];
            a[mt].q[1] = *(const intx4*)&As[m][16 + kb];
        }
        // B fragment: lane n = l&15, two 8-elem K runs selected by l>>4
        const int kh = (lane >> 4) * 8;
        #pragma unroll
        for (int nt = 0; nt < 4; ++nt) {
            int n = 16 * nt + (lane & 15);
            b[nt].q[0] = *(const intx4*)&Bs[n][kh];
            b[nt].q[1] = *(const intx4*)&Bs[n][16 + kh];
        }
        #pragma unroll
        for (int mt = 0; mt < 2; ++mt)
            #pragma unroll
            for (int nt = 0; nt < 4; ++nt)
                acc[mt][nt] = __builtin_amdgcn_wmma_f32_16x16x32_bf16(
                    false, a[mt].v, false, b[nt].v,
                    (short)0, acc[mt][nt], false, false);
        __syncthreads();
    }

    // Epilogue: C layout -> lane row m = r + 8*(l>>4), col n = l&15
    #pragma unroll
    for (int mt = 0; mt < 2; ++mt) {
        #pragma unroll
        for (int nt = 0; nt < 4; ++nt) {
            #pragma unroll
            for (int r = 0; r < 8; ++r) {
                int m = m0 + wrow + 16 * mt + r + 8 * (lane >> 4);
                int n = n0 + 16 * nt + (lane & 15);
                float v = acc[mt][nt][r];
                if (OUT_MODE == 2) {
                    Yf[(size_t)m * N + n] = v + bias[n];
                } else if (OUT_MODE == 1) {
                    int bb = m >> 11, s = m & (SEQ - 1);
                    int h  = n >> 6,  d = n & (HEAD_DIM - 1);
                    Ybf[((size_t)(bb * NUM_HEADS + h) * SEQ + s) * HEAD_DIM + d] =
                        f2bf(v);
                } else {           // OUT_MODE 3: V pre-transposed [B,H,D,S]
                    int bb = m >> 11, s = m & (SEQ - 1);
                    int h  = n >> 6,  d = n & (HEAD_DIM - 1);
                    Ybf[((size_t)(bb * NUM_HEADS + h) * HEAD_DIM + d) * SEQ + s] =
                        f2bf(v);
                }
            }
        }
    }
}

// ---------------------------------------------------------------------------
// Flash attention: one workgroup per (batch, head, 64-row Q block).
// 4 waves, each owns a 16-row strip; online softmax over 64-key blocks.
// Qg/Kg: [B,H,S,64] bf16    Vg: [B,H,64,S] bf16 (pre-transposed)
// ---------------------------------------------------------------------------
__global__ void __launch_bounds__(128)
flash_attn_wmma(const unsigned short* __restrict__ Qg,
                const unsigned short* __restrict__ Kg,
                const unsigned short* __restrict__ Vg,
                unsigned short* __restrict__ Og)         // [B,S,H*64] bf16
{
    constexpr int PAD = 8;
    __shared__ unsigned short Qs[64][HEAD_DIM + PAD];    // [row][dim]
    __shared__ unsigned short Ks[64][HEAD_DIM + PAD];    // [key][dim]
    __shared__ unsigned short Vt[HEAD_DIM][64 + PAD];    // [dim][key]
    __shared__ unsigned short Ps[64][64 + PAD];          // [row][key] probs bf16

    const int t    = threadIdx.x;
    const int lane = t & 31;
    const int wave = t >> 5;
    const int qb   = blockIdx.x * 64;
    const int h    = blockIdx.y;
    const int bi   = blockIdx.z;
    const size_t headBase = (size_t)(bi * NUM_HEADS + h) * SEQ * HEAD_DIM;
    const int wbase = wave * 16;

    // Stage Q tile once: 64 x 64 bf16 = 512 x 16B, 4 per thread
    #pragma unroll
    for (int i = 0; i < 4; ++i) {
        int idx = t + 128 * i;
        int r = idx >> 3, c = idx & 7;
        stage16(Qg + headBase + (size_t)(qb + r) * HEAD_DIM + c * 8,
                &Qs[r][c * 8]);
    }

    floatx8 oacc[4];
    #pragma unroll
    for (int i = 0; i < 4; ++i) oacc[i] = 0.0f;
    float mstate[8], lstate[8];
    #pragma unroll
    for (int r = 0; r < 8; ++r) { mstate[r] = -3.0e38f; lstate[r] = 0.0f; }

    const float scale = 0.125f;                    // 1/sqrt(HEAD_DIM)
    const int kbq = (lane >> 4) * 8;               // A-frag K split
    const int kh  = (lane >> 4) * 8;               // B-frag K split

    for (int kb0 = 0; kb0 < SEQ; kb0 += 64) {
        __syncthreads();
        // Stage K tile [key][dim] and Vt tile [dim][key] — both are plain
        // row-contiguous 16B copies now that V is pre-transposed in memory.
        #pragma unroll
        for (int i = 0; i < 4; ++i) {
            int idx = t + 128 * i;
            int r = idx >> 3, c = idx & 7;
            stage16(Kg + headBase + (size_t)(kb0 + r) * HEAD_DIM + c * 8,
                    &Ks[r][c * 8]);
            stage16(Vg + headBase + (size_t)r * SEQ + kb0 + c * 8,
                    &Vt[r][c * 8]);
        }
        wait_stage();
        __syncthreads();

        // S = Q @ K^T  (M=16 rows, N=64 keys, K=64 dims in 2 steps)
        floatx8 sacc[4];
        #pragma unroll
        for (int i = 0; i < 4; ++i) sacc[i] = 0.0f;
        #pragma unroll
        for (int ks = 0; ks < 2; ++ks) {
            FragA a, b[4];
            int m = wbase + (lane & 15);
            a.q[0] = *(const intx4*)&Qs[m][ks * 32 + kbq];
            a.q[1] = *(const intx4*)&Qs[m][ks * 32 + 16 + kbq];
            #pragma unroll
            for (int nt = 0; nt < 4; ++nt) {       // hoist all B fragments
                int n = 16 * nt + (lane & 15);
                b[nt].q[0] = *(const intx4*)&Ks[n][ks * 32 + kh];
                b[nt].q[1] = *(const intx4*)&Ks[n][ks * 32 + 16 + kh];
            }
            #pragma unroll
            for (int nt = 0; nt < 4; ++nt)
                sacc[nt] = __builtin_amdgcn_wmma_f32_16x16x32_bf16(
                    false, a.v, false, b[nt].v, (short)0, sacc[nt], false, false);
        }

        // Online softmax. C layout: VGPR r holds row r (lanes 0-15) / r+8
        // (lanes 16-31); row reduction = shfl_xor within 16-lane groups.
        #pragma unroll
        for (int r = 0; r < 8; ++r) {
            float s0 = sacc[0][r] * scale, s1 = sacc[1][r] * scale;
            float s2 = sacc[2][r] * scale, s3 = sacc[3][r] * scale;
            float pm = fmaxf(fmaxf(s0, s1), fmaxf(s2, s3));
            #pragma unroll
            for (int off = 1; off < 16; off <<= 1)
                pm = fmaxf(pm, __shfl_xor(pm, off, 32));
            float mn    = fmaxf(mstate[r], pm);
            float alpha = __expf(mstate[r] - mn);
            mstate[r]   = mn;
            float p0 = __expf(s0 - mn), p1 = __expf(s1 - mn);
            float p2 = __expf(s2 - mn), p3 = __expf(s3 - mn);
            float ps = p0 + p1 + p2 + p3;
            #pragma unroll
            for (int off = 1; off < 16; off <<= 1)
                ps += __shfl_xor(ps, off, 32);
            lstate[r] = lstate[r] * alpha + ps;
            #pragma unroll
            for (int nt = 0; nt < 4; ++nt) oacc[nt][r] *= alpha;
            int row = wbase + r + 8 * (lane >> 4);
            int nc  = lane & 15;
            Ps[row][ 0 + nc] = f2bf(p0);
            Ps[row][16 + nc] = f2bf(p1);
            Ps[row][32 + nc] = f2bf(p2);
            Ps[row][48 + nc] = f2bf(p3);
        }

        // O += P @ V  (A = this wave's Ps rows, B = Vt[dim][key])
        #pragma unroll
        for (int ks = 0; ks < 2; ++ks) {
            FragA a, b[4];
            int m = wbase + (lane & 15);
            a.q[0] = *(const intx4*)&Ps[m][ks * 32 + kbq];
            a.q[1] = *(const intx4*)&Ps[m][ks * 32 + 16 + kbq];
            #pragma unroll
            for (int nt = 0; nt < 4; ++nt) {       // hoist all B fragments
                int n = 16 * nt + (lane & 15);
                b[nt].q[0] = *(const intx4*)&Vt[n][ks * 32 + kh];
                b[nt].q[1] = *(const intx4*)&Vt[n][ks * 32 + 16 + kh];
            }
            #pragma unroll
            for (int nt = 0; nt < 4; ++nt)
                oacc[nt] = __builtin_amdgcn_wmma_f32_16x16x32_bf16(
                    false, a.v, false, b[nt].v, (short)0, oacc[nt], false, false);
        }
    }

    // Normalize and write O in [B,S,H*64] layout (bf16 for final projection)
    #pragma unroll
    for (int r = 0; r < 8; ++r) {
        float inv = 1.0f / lstate[r];
        int row = qb + wbase + r + 8 * (lane >> 4);
        #pragma unroll
        for (int nt = 0; nt < 4; ++nt) {
            int d = 16 * nt + (lane & 15);
            Og[(size_t)(bi * SEQ + row) * D_MODEL + h * HEAD_DIM + d] =
                f2bf(oacc[nt][r] * inv);
        }
    }
}

// ---------------------------------------------------------------------------
// Host launcher
// ---------------------------------------------------------------------------
extern "C" void kernel_launch(void* const* d_in, const int* in_sizes, int n_in,
                              void* d_out, int out_size, void* d_ws, size_t ws_size,
                              hipStream_t stream) {
    const float* q_in = (const float*)d_in[0];
    const float* k_in = (const float*)d_in[1];
    const float* v_in = (const float*)d_in[2];
    const float* Wq   = (const float*)d_in[3];
    const float* Wk   = (const float*)d_in[4];
    const float* Wv   = (const float*)d_in[5];
    const float* Wo   = (const float*)d_in[6];
    const float* bo   = (const float*)d_in[7];
    float* out = (float*)d_out;

    const int M = BATCH * SEQ;                           // 4096
    const size_t SZ_ACT = (size_t)M * D_MODEL * 2;       // bf16 activation: 8MB
    const size_t SZ_W   = (size_t)D_MODEL * D_MODEL * 2; // bf16 weight: 2MB

    char* ws = (char*)d_ws;                              // total ~64MB scratch
    unsigned short* qbf = (unsigned short*)(ws + 0 * SZ_ACT);
    unsigned short* kbf = (unsigned short*)(ws + 1 * SZ_ACT);
    unsigned short* vbf = (unsigned short*)(ws + 2 * SZ_ACT);
    unsigned short* wqb = (unsigned short*)(ws + 3 * SZ_ACT + 0 * SZ_W);
    unsigned short* wkb = (unsigned short*)(ws + 3 * SZ_ACT + 1 * SZ_W);
    unsigned short* wvb = (unsigned short*)(ws + 3 * SZ_ACT + 2 * SZ_W);
    unsigned short* wob = (unsigned short*)(ws + 3 * SZ_ACT + 3 * SZ_W);
    unsigned short* Qb  = (unsigned short*)(ws + 3 * SZ_ACT + 4 * SZ_W + 0 * SZ_ACT);
    unsigned short* Kb  = (unsigned short*)(ws + 3 * SZ_ACT + 4 * SZ_W + 1 * SZ_ACT);
    unsigned short* Vb  = (unsigned short*)(ws + 3 * SZ_ACT + 4 * SZ_W + 2 * SZ_ACT);
    unsigned short* Ob  = (unsigned short*)(ws + 3 * SZ_ACT + 4 * SZ_W + 3 * SZ_ACT);

    // 1) Convert everything to bf16
    const int nact4 = M * D_MODEL / 4;                   // 1,048,576
    const int nw4   = D_MODEL * D_MODEL / 4;             // 262,144
    cvt_f32_bf16<<<(nact4 + 255) / 256, 256, 0, stream>>>(q_in, qbf, nact4);
    cvt_f32_bf16<<<(nact4 + 255) / 256, 256, 0, stream>>>(k_in, kbf, nact4);
    cvt_f32_bf16<<<(nact4 + 255) / 256, 256, 0, stream>>>(v_in, vbf, nact4);
    cvt_f32_bf16<<<(nw4   + 255) / 256, 256, 0, stream>>>(Wq, wqb, nw4);
    cvt_f32_bf16<<<(nw4   + 255) / 256, 256, 0, stream>>>(Wk, wkb, nw4);
    cvt_f32_bf16<<<(nw4   + 255) / 256, 256, 0, stream>>>(Wv, wvb, nw4);
    cvt_f32_bf16<<<(nw4   + 255) / 256, 256, 0, stream>>>(Wo, wob, nw4);

    // 2) Q/K/V projections
    dim3 ggrid(M / 128, D_MODEL / 64);                   // (32, 16)
    gemm_wmma_bf16<1><<<ggrid, 128, 0, stream>>>(qbf, wqb, Qb, nullptr, nullptr,
                                                 M, D_MODEL, D_MODEL);
    gemm_wmma_bf16<1><<<ggrid, 128, 0, stream>>>(kbf, wkb, Kb, nullptr, nullptr,
                                                 M, D_MODEL, D_MODEL);
    gemm_wmma_bf16<3><<<ggrid, 128, 0, stream>>>(vbf, wvb, Vb, nullptr, nullptr,
                                                 M, D_MODEL, D_MODEL);

    // 3) Flash attention per (q-block, head, batch)
    dim3 agrid(SEQ / 64, NUM_HEADS, BATCH);              // (32, 16, 2)
    flash_attn_wmma<<<agrid, 128, 0, stream>>>(Qb, Kb, Vb, Ob);

    // 4) Output projection with bias -> f32 d_out
    gemm_wmma_bf16<2><<<ggrid, 128, 0, stream>>>(Ob, wob, nullptr, out, bo,
                                                 M, D_MODEL, D_MODEL);
}
